// MultipleHeadAttention_5317169512775
// MI455X (gfx1250) — matrix operand order repaired and checked by reading, not compiled, not run on hardware
//
#include <hip/hip_runtime.h>
#include <hip/hip_bf16.h>

// ---------------------------------------------------------------------------
// MI455X (gfx1250) implementation.
//  - All heavy GEMMs (Wq/Wk/Wv proj, FFN1, FFN2) via v_wmma_f32_16x16x32_bf16,
//    f32 accumulation, LDS-tiled 128x128x32, 8 waves/block (wave32).
//  - Attention (per b,n,h) also via WMMA, fragments loaded straight from
//    global as 16B vectors (projection epilogues write attention-native
//    layouts).
//  - FFN GEMM X operands are stored pre-transposed [b][m][c] bf16 so the
//    LDS X-tile staging is contiguous 16B row copies done with the gfx1250
//    async global->LDS DMA path (global_load_async_to_lds_b128 + ASYNCcnt).
//  - f32->bf16 staging uses packed pair conversion + uint4 LDS stores.
// ---------------------------------------------------------------------------

typedef __attribute__((ext_vector_type(16))) __bf16 v16bf;
typedef __attribute__((ext_vector_type(8)))  float  v8f;

static constexpr int   Bq     = 2;
static constexpr int   C      = 384;
static constexpr int   Npts   = 2048;
static constexpr int   Kn     = 32;              // neighbors == query count
static constexpr int   H      = 6;
static constexpr int   DH     = 64;
static constexpr int   NM     = Npts * Kn;       // 65536 columns per batch
static constexpr float BN_EPS = 1e-5f;

union FragB16 {
  v16bf v;
  uint4 q[2];
};

__device__ __forceinline__ v8f wmma_bf16(const FragB16& a, const FragB16& b, v8f c) {
  // D = A(16x32 bf16) * B(32x16 bf16) + C(16x16 f32)
  return __builtin_amdgcn_wmma_f32_16x16x32_bf16(false, a.v, false, b.v,
                                                 (short)0, c, false, false);
}

__device__ __forceinline__ unsigned pack_bf16(float lo, float hi) {
  union { __bf16 b[2]; unsigned u; } t;
  t.b[0] = (__bf16)lo;
  t.b[1] = (__bf16)hi;
  return t.u;
}

// ---------------------------------------------------------------------------
// Generic GEMM:  OUT[b][o][m] = sum_c W[o][c] * X[b][c][m]  (+ fused epilogue)
// MODE 0: X f32 [b][c][m];  out bf16 layout [B,N,H,Qc,DH]   (Q and K)
// MODE 2: X f32 [b][c][m];  out bf16 layout [B,N,H,DH,K]    (V)
// MODE 3: X bf16 [b][m][c] (transposed); (acc+b1)*bnScale+bnShift, ReLU
//         -> bf16 transposed [b][m][o]
// MODE 4: X bf16 [b][m][c] (transposed); acc+b2+res, ReLU -> f32 [b][o][m]
// ---------------------------------------------------------------------------
template <int MODE>
__global__ __launch_bounds__(256) void gemm_fused_kernel(
    const float* __restrict__ Wm,    // [COUT, CIN] f32
    const void*  __restrict__ Xsrc,  // see per-mode layouts above
    const float* __restrict__ bias,  // [COUT]
    const float* __restrict__ gamma, const float* __restrict__ beta,
    const float* __restrict__ rmean, const float* __restrict__ rvar,
    const float* __restrict__ resF,  // [B, C, NM] f32 residual (MODE 4)
    void* __restrict__ Out)
{
  constexpr bool XF32 = (MODE < 3);
  constexpr int  CIN  = (MODE == 4) ? 2 * C : C;
  constexpr int  COUT = (MODE == 3) ? 2 * C : C;

  __shared__ __align__(16) __bf16 sW[128 * 32];  // [o_local][k]   8 KB
  __shared__ __align__(16) __bf16 sX[128 * 32];  // [m_local][k]   8 KB

  const int tid = threadIdx.x;
  const int w   = tid >> 5;          // wave 0..7
  const int l   = tid & 31;
  const int lh  = l >> 4;            // half-wave select
  const int ll  = l & 15;
  const int wm  = w >> 1;            // 0..3  -> 32-row strip
  const int wn  = w & 1;             // 0..1  -> 64-col strip

  const int m0 = blockIdx.x * 128;
  const int o0 = blockIdx.y * 128;
  const int bz = blockIdx.z;

  const float*  Xf  = (const float*)Xsrc  + (size_t)bz * CIN * NM;  // [c][m]
  const __bf16* XTb = (const __bf16*)Xsrc + (size_t)bz * NM * CIN;  // [m][c]

  v8f acc[2][4];
  const v8f vzero = {0.f, 0.f, 0.f, 0.f, 0.f, 0.f, 0.f, 0.f};
#pragma unroll
  for (int i = 0; i < 2; ++i)
#pragma unroll
    for (int j = 0; j < 4; ++j) acc[i][j] = vzero;

  for (int kt = 0; kt < CIN / 32; ++kt) {
    // ---- prefetch next k-tile into L2 (global_prefetch) ----
    if (kt + 1 < CIN / 32) {
      __builtin_prefetch(Wm + (size_t)(o0 + (tid >> 1)) * CIN + (kt + 1) * 32, 0, 3);
      if constexpr (XF32)
        __builtin_prefetch(
            Xf + (size_t)((kt + 1) * 32 + (tid & 15) * 2) * NM + m0 + (tid >> 4) * 8,
            0, 3);
    }

    // ---- stage W tile [128][32]: packed bf16 pairs, uint4 LDS stores ----
    {
      const int row   = tid >> 1;
      const int halfc = (tid & 1) * 16;
      const float* wp = Wm + (size_t)(o0 + row) * CIN + kt * 32 + halfc;
      uint4 u0, u1;
      u0.x = pack_bf16(wp[0],  wp[1]);   u0.y = pack_bf16(wp[2],  wp[3]);
      u0.z = pack_bf16(wp[4],  wp[5]);   u0.w = pack_bf16(wp[6],  wp[7]);
      u1.x = pack_bf16(wp[8],  wp[9]);   u1.y = pack_bf16(wp[10], wp[11]);
      u1.z = pack_bf16(wp[12], wp[13]);  u1.w = pack_bf16(wp[14], wp[15]);
      *(uint4*)&sW[row * 32 + halfc]     = u0;
      *(uint4*)&sW[row * 32 + halfc + 8] = u1;
    }

    // ---- stage X tile -> sX[m_local][k] ----
    if constexpr (XF32) {
      // f32 [c][m] source: transpose while converting; paired c-columns so
      // each LDS store is a packed bf16x2.
      const int cc = (tid & 15) * 2;     // 0,2,...,30
      const int mm = (tid >> 4) * 8;     // 0,8,...,120
      const float* x0 = Xf + (size_t)(kt * 32 + cc) * NM + m0 + mm;
      const float* x1 = x0 + NM;
#pragma unroll
      for (int i = 0; i < 8; ++i)
        *(unsigned*)&sX[(mm + i) * 32 + cc] = pack_bf16(x0[i], x1[i]);
    } else {
      // bf16 transposed [m][c] source: pure 16B row copies via gfx1250
      // async global->LDS DMA (tracked by ASYNCcnt).
#pragma unroll
      for (int rep = 0; rep < 2; ++rep) {
        const int idx  = rep * 256 + tid;      // 512 chunks of 16 B
        const int row  = idx >> 2;             // 0..127
        const int part = (idx & 3) * 8;        // bf16 elements: 0,8,16,24
        const __bf16* src = XTb + (size_t)(m0 + row) * CIN + kt * 32 + part;
        const unsigned ldsAddr = (unsigned)(size_t)&sX[row * 32 + part];
        asm volatile("global_load_async_to_lds_b128 %0, %1, off"
                     :
                     : "v"(ldsAddr), "v"((unsigned long long)(size_t)src)
                     : "memory");
      }
      asm volatile("s_wait_asynccnt 0x0" ::: "memory");
    }
    __syncthreads();

    // ---- fragments + 8 WMMAs ----
    FragB16 afr[2];
#pragma unroll
    for (int mi = 0; mi < 2; ++mi) {
      const int orow = wm * 32 + mi * 16 + ll;
      const int koff = lh * 16;                       // A: K 0-15 / 16-31
      afr[mi].q[0] = *(const uint4*)&sW[orow * 32 + koff];
      afr[mi].q[1] = *(const uint4*)&sW[orow * 32 + koff + 8];
    }
#pragma unroll
    for (int nj = 0; nj < 4; ++nj) {
      const int ncol = wn * 64 + nj * 16 + ll;
      const int k0   = lh * 16;                       // B: K 0-15 / 16-31
      FragB16 bfr;
      bfr.q[0] = *(const uint4*)&sX[ncol * 32 + k0];
      bfr.q[1] = *(const uint4*)&sX[ncol * 32 + k0 + 8];
      acc[0][nj] = wmma_bf16(afr[0], bfr, acc[0][nj]);
      acc[1][nj] = wmma_bf16(afr[1], bfr, acc[1][nj]);
    }
    __syncthreads();
  }

  // ---- fused epilogue ----
#pragma unroll
  for (int mi = 0; mi < 2; ++mi) {
#pragma unroll
    for (int r = 0; r < 8; ++r) {
      const int o = o0 + wm * 32 + mi * 16 + lh * 8 + r;   // D: vgpr r -> M r/(r+8)
      float bnScale = 0.f, bnShift = 0.f, bo = 0.f;
      if constexpr (MODE == 3) {
        bnScale = gamma[o] * rsqrtf(rvar[o] + BN_EPS);
        bnShift = beta[o] - rmean[o] * bnScale + bias[o] * bnScale;
      } else {
        bo = bias[o];
      }
#pragma unroll
      for (int nj = 0; nj < 4; ++nj) {
        const int m = m0 + wn * 64 + nj * 16 + ll;
        float val = acc[mi][nj][r];
        if constexpr (MODE == 0) {            // Q/K layout [B,N,H,Qc,DH]
          const int n_pt = m >> 5, qc = m & 31;
          const int h = o >> 6, d = o & 63;
          const size_t off =
              ((((size_t)bz * Npts + n_pt) * H + h) * Kn + qc) * DH + d;
          ((__bf16*)Out)[off] = (__bf16)(val + bo);
        } else if constexpr (MODE == 2) {     // V layout [B,N,H,DH,K]
          const int n_pt = m >> 5, kc = m & 31;
          const int h = o >> 6, d = o & 63;
          const size_t off =
              ((((size_t)bz * Npts + n_pt) * H + h) * DH + d) * Kn + kc;
          ((__bf16*)Out)[off] = (__bf16)(val + bo);
        } else if constexpr (MODE == 3) {     // BN(eval)+ReLU -> bf16 [b][m][o]
          val = val * bnScale + bnShift;
          val = fmaxf(val, 0.f);
          ((__bf16*)Out)[((size_t)bz * NM + m) * COUT + o] = (__bf16)val;
        } else {                              // FFN2: +b2 +res, ReLU -> f32
          const size_t idx = ((size_t)bz * COUT + o) * NM + m;
          val = fmaxf(val + bo + resF[idx], 0.f);
          ((float*)Out)[idx] = val;
        }
      }
    }
  }
}

// ---------------------------------------------------------------------------
// Attention: one block per (b,n), one wave per head.
//   logits = (q @ k) * DH^-1/2 ; softmax over K ; agg = attn @ v + residual
// ---------------------------------------------------------------------------
__global__ __launch_bounds__(192) void attn_kernel(
    const __bf16* __restrict__ Qt,      // [B,N,H,Qc,DH]
    const __bf16* __restrict__ Kp,      // [B,N,H,Kc,DH]
    const __bf16* __restrict__ Vv,      // [B,N,H,DH,Kc]
    const float*  __restrict__ queries, // [B,C,N,Kn] f32 residual
    float*  __restrict__ aggF,          // [B,C,N,Kn] f32
    __bf16* __restrict__ aggB)          // transposed [B,NM,C] bf16
{
  __shared__ __align__(16) float  sLog[H][Kn * Kn];  // 24 KB
  __shared__ __align__(16) __bf16 sAtt[H][Kn * Kn];  // 12 KB

  const int tid = threadIdx.x;
  const int h   = tid >> 5;            // head == wave
  const int l   = tid & 31;
  const int lh  = l >> 4;
  const int ll  = l & 15;
  const int bn  = blockIdx.x;
  const int b   = bn / Npts;
  const int n   = bn % Npts;

  const size_t  headBase = ((size_t)b * Npts + n) * H + h;
  const __bf16* qbase = Qt + headBase * (Kn * DH);   // [qc][d]
  const __bf16* kbase = Kp + headBase * (Kn * DH);   // [kc][d]
  const __bf16* vbase = Vv + headBase * (DH * Kn);   // [d][kc]

  const v8f vzero = {0.f, 0.f, 0.f, 0.f, 0.f, 0.f, 0.f, 0.f};

  // ---- logits: 32x32 = q[32x64] @ k^T, 2 k-steps of 32 ----
  v8f accL[2][2];
#pragma unroll
  for (int i = 0; i < 2; ++i)
#pragma unroll
    for (int j = 0; j < 2; ++j) accL[i][j] = vzero;

#pragma unroll
  for (int s = 0; s < 2; ++s) {
    FragB16 qa[2];
#pragma unroll
    for (int mi = 0; mi < 2; ++mi) {
      const int qc   = mi * 16 + ll;
      const int koff = lh * 16;
      const __bf16* p = qbase + qc * DH + s * 32 + koff;
      qa[mi].q[0] = *(const uint4*)p;
      qa[mi].q[1] = *(const uint4*)(p + 8);
    }
#pragma unroll
    for (int nj = 0; nj < 2; ++nj) {
      const int kc = nj * 16 + ll;
      const int k0 = lh * 16;
      const __bf16* p = kbase + kc * DH + s * 32 + k0;
      FragB16 kb;
      kb.q[0] = *(const uint4*)p;
      kb.q[1] = *(const uint4*)(p + 8);
      accL[0][nj] = wmma_bf16(qa[0], kb, accL[0][nj]);
      accL[1][nj] = wmma_bf16(qa[1], kb, accL[1][nj]);
    }
  }

  const float scale = 0.125f;  // 64^-0.5
#pragma unroll
  for (int mi = 0; mi < 2; ++mi)
#pragma unroll
    for (int nj = 0; nj < 2; ++nj)
#pragma unroll
      for (int r = 0; r < 8; ++r) {
        const int row = mi * 16 + lh * 8 + r;
        const int col = nj * 16 + ll;
        sLog[h][row * 32 + col] = accL[mi][nj][r] * scale;
      }
  __syncthreads();

  // ---- softmax: one row per lane ----
  {
    const float* rp = &sLog[h][l * 32];
    float mx = -1e30f;
#pragma unroll
    for (int j = 0; j < 32; ++j) mx = fmaxf(mx, rp[j]);
    float sum = 0.f;
#pragma unroll
    for (int j = 0; j < 32; ++j) sum += __expf(rp[j] - mx);
    const float inv = 1.f / sum;
#pragma unroll
    for (int j = 0; j < 16; ++j)
      *(unsigned*)&sAtt[h][l * 32 + 2 * j] =
          pack_bf16(__expf(rp[2 * j] - mx) * inv, __expf(rp[2 * j + 1] - mx) * inv);
  }
  __syncthreads();

  // ---- agg: 32x64 = attn[32x32] @ v[32x64], single k-step ----
  v8f accA[2][4];
#pragma unroll
  for (int i = 0; i < 2; ++i)
#pragma unroll
    for (int j = 0; j < 4; ++j) accA[i][j] = vzero;

  FragB16 aa[2];
#pragma unroll
  for (int mi = 0; mi < 2; ++mi) {
    const int qc   = mi * 16 + ll;
    const int koff = lh * 16;
    const __bf16* p = &sAtt[h][qc * 32 + koff];
    aa[mi].q[0] = *(const uint4*)p;
    aa[mi].q[1] = *(const uint4*)(p + 8);
  }
#pragma unroll
  for (int nj = 0; nj < 4; ++nj) {
    const int d  = nj * 16 + ll;
    const int k0 = lh * 16;
    const __bf16* p = vbase + d * 32 + k0;
    FragB16 vb;
    vb.q[0] = *(const uint4*)p;
    vb.q[1] = *(const uint4*)(p + 8);
    accA[0][nj] = wmma_bf16(aa[0], vb, accA[0][nj]);
    accA[1][nj] = wmma_bf16(aa[1], vb, accA[1][nj]);
  }

  // ---- epilogue: merge heads, + residual, dual-precision store ----
#pragma unroll
  for (int mi = 0; mi < 2; ++mi)
#pragma unroll
    for (int nj = 0; nj < 4; ++nj)
#pragma unroll
      for (int r = 0; r < 8; ++r) {
        const int qc = mi * 16 + lh * 8 + r;
        const int d  = nj * 16 + ll;
        const int c  = h * DH + d;
        const size_t idx = (((size_t)b * C + c) * Npts + n) * Kn + qc;
        const float v = accA[mi][nj][r] + queries[idx];
        aggF[idx] = v;
        // transposed bf16 copy [b][m][c] for async-staged FFN1 GEMM
        aggB[((size_t)b * NM + n * Kn + qc) * C + c] = (__bf16)v;
      }
}

// ---------------------------------------------------------------------------
extern "C" void kernel_launch(void* const* d_in, const int* in_sizes, int n_in,
                              void* d_out, int out_size, void* d_ws, size_t ws_size,
                              hipStream_t stream) {
  const float* queries = (const float*)d_in[0];
  const float* feats   = (const float*)d_in[1];
  const float* Wq = (const float*)d_in[2];
  const float* bq = (const float*)d_in[3];
  const float* Wk = (const float*)d_in[4];
  const float* bk = (const float*)d_in[5];
  const float* Wv = (const float*)d_in[6];
  const float* bv = (const float*)d_in[7];
  const float* W1 = (const float*)d_in[8];
  const float* b1 = (const float*)d_in[9];
  const float* gamma = (const float*)d_in[10];
  const float* beta  = (const float*)d_in[11];
  const float* rmean = (const float*)d_in[12];
  const float* rvar  = (const float*)d_in[13];
  const float* W2 = (const float*)d_in[14];
  const float* b2 = (const float*)d_in[15];

  // Workspace carve-up (~806 MB total).
  char* ws = (char*)d_ws;
  const size_t nElem = (size_t)Bq * C * Npts * Kn;  // 50,331,648
  __bf16* Qt   = (__bf16*)ws;  ws += nElem * 2;
  __bf16* Kp   = (__bf16*)ws;  ws += nElem * 2;
  __bf16* Vv   = (__bf16*)ws;  ws += nElem * 2;
  __bf16* aggB = (__bf16*)ws;  ws += nElem * 2;     // transposed [B,NM,C]
  float*  aggF = (float*)ws;   ws += nElem * 4;
  __bf16* hB   = (__bf16*)ws;  ws += nElem * 2 * 2; // transposed [B,NM,2C]

  const dim3 blk(256);
  const dim3 gProj(NM / 128, C / 128, Bq);       // 512 x 3 x 2
  const dim3 gFfn1(NM / 128, (2 * C) / 128, Bq); // 512 x 6 x 2

  // QKV projections (bf16 WMMA, attention-native output layouts)
  gemm_fused_kernel<0><<<gProj, blk, 0, stream>>>(
      Wq, queries, bq, nullptr, nullptr, nullptr, nullptr, nullptr, Qt);
  gemm_fused_kernel<0><<<gProj, blk, 0, stream>>>(
      Wk, feats, bk, nullptr, nullptr, nullptr, nullptr, nullptr, Kp);
  gemm_fused_kernel<2><<<gProj, blk, 0, stream>>>(
      Wv, feats, bv, nullptr, nullptr, nullptr, nullptr, nullptr, Vv);

  // Attention + residual (WMMA per head)
  attn_kernel<<<dim3(Bq * Npts), dim3(192), 0, stream>>>(
      Qt, Kp, Vv, queries, aggF, aggB);

  // FFN1: conv1x1 -> BN(eval) -> ReLU  (async-staged X, fused epilogue)
  gemm_fused_kernel<3><<<gFfn1, blk, 0, stream>>>(
      W1, aggB, b1, gamma, beta, rmean, rvar, nullptr, hB);

  // FFN2: conv1x1 + residual + ReLU -> f32 output (async-staged X)
  gemm_fused_kernel<4><<<gProj, blk, 0, stream>>>(
      W2, hB, b2, nullptr, nullptr, nullptr, nullptr, aggF, d_out);
}
